// QueryMaskedAttention_40681930228038
// MI455X (gfx1250) — matrix operand order repaired and checked by reading, not compiled, not run on hardware
//
#include <hip/hip_runtime.h>

// ---------------------------------------------------------------------------
// Problem constants (from reference): B=8, C=64, N=4096, K=16
// ---------------------------------------------------------------------------
#define BATCH    8
#define CHN      64
#define NPOINT   4096
#define KNBR     16
#define NKTOT    (NPOINT * KNBR)              // 65536 positions per (b, channel)
#define TILES_TOTAL ((BATCH * NKTOT) / 16)    // 32768 position tiles of 16
#define WAVES_TOTAL 8192
#define TILES_PER_WAVE (TILES_TOTAL / WAVES_TOTAL)  // 4
#define NBLOCKS  (WAVES_TOTAL * 32 / 256)     // 1024 blocks of 256 threads
#define PSTRIDE  36                           // padded dwords per lane (LDS)

typedef __attribute__((ext_vector_type(16))) __bf16 v16bf;
typedef __attribute__((ext_vector_type(8)))  float  v8f;

// D = A(16x32 bf16) x B(32x16 bf16) + C(16x16 f32), wave32 WMMA
static __device__ __forceinline__ v8f wmma_bf(v16bf a, v16bf b, v8f c) {
  return __builtin_amdgcn_wmma_f32_16x16x32_bf16(false, a, false, b, (short)0, c,
                                                 false, false);
}

// Exchange with lane^16 (SWAPX16 ds_swizzle pattern: xor=0x10, and=0x1f)
static __device__ __forceinline__ float swapx16(float x) {
  return __int_as_float(__builtin_amdgcn_ds_swizzle(__float_as_int(x), 0x401F));
}

// Convert two adjacent f32 D-tiles (rows 32j..32j+31 of a 64-channel result)
// into one bf16 B-operand (channels 32j..32j+31 x 16 positions).
// D layout: lanes 0-15 hold rows 0..7 (VGPR r), lanes 16-31 hold rows 8..15.
// B layout: lanes 0-15 hold K 0..15 (elem e), lanes 16-31 hold K 16..31.
static __device__ __forceinline__ v16bf relayout(v8f d0, v8f d1, int upper) {
  v16bf b;
#pragma unroll
  for (int r = 0; r < 8; ++r) {
    float send = upper ? d0[r] : d1[r];   // value the partner half needs
    float keep = upper ? d1[r] : d0[r];   // value this half keeps
    float part = swapx16(send);
    float lo = upper ? part : keep;       // rows 0..7  of this half's tile
    float hi = upper ? keep : part;       // rows 8..15 of this half's tile
    b[r]     = (__bf16)lo;
    b[r + 8] = (__bf16)hi;
  }
  return b;
}

// Stage conv weights into LDS, bf16, pre-swizzled into the A-operand lane
// layout: per (conv, o-tile t, c-chunk j, lane) a contiguous 32B (16 bf16).
// A elem e (lane half h): c = 32j + 16h + e.
static __device__ __forceinline__ void stage_weights(
    __bf16* ldsW, float* ldsB, const float* const* Ws, const float* const* Bsv,
    const int* cins, int nconv) {
  for (int idx = threadIdx.x; idx < nconv * 2048; idx += blockDim.x) {
    int conv  = idx >> 11;
    int rem   = idx & 2047;
    int entry = rem >> 3;        // (t*2+j)*32 + lane
    int d     = rem & 7;         // dword (pair of bf16) within lane's 32B
    int L     = entry & 31;
    int tj    = entry >> 5;
    int t     = tj >> 1;
    int j     = tj & 1;
    int o     = t * 16 + (L & 15);
    int half  = L >> 4;
    int base0 = 32 * j + 16 * half;
    int c0    = base0 + 2 * d;
    int cin   = cins[conv];
    const float* W = Ws[conv];
    float w0 = (c0     < cin) ? W[o * cin + c0]     : 0.f;
    float w1 = (c0 + 1 < cin) ? W[o * cin + c0 + 1] : 0.f;
    ldsW[2 * idx]     = (__bf16)w0;
    ldsW[2 * idx + 1] = (__bf16)w1;
  }
  for (int i = threadIdx.x; i < nconv * 64; i += blockDim.x)
    ldsB[i] = Bsv[i >> 6][i & 63];
}

// One 64-out-channel conv for 16 positions: 4 o-tiles x NCHUNK c-chunks of
// WMMA. Accumulators are initialized with the bias (added exactly once).
template <int NCHUNK>
static __device__ __forceinline__ void conv_apply(const __bf16* ldsW,
                                                  const float* ldsB, int conv,
                                                  int lane, const v16bf* bop,
                                                  v8f out[4]) {
  const int half = lane >> 4;
#pragma unroll
  for (int t = 0; t < 4; ++t) {
    v8f acc;
    const float* bp = ldsB + conv * 64 + t * 16 + half * 8;
#pragma unroll
    for (int r = 0; r < 8; ++r) acc[r] = bp[r];
#pragma unroll
    for (int j = 0; j < NCHUNK; ++j) {
      const v16bf* A =
          (const v16bf*)(ldsW + ((conv * 8 + t * 2 + j) * 32 + lane) * 16);
      acc = wmma_bf(*A, bop[j], acc);
    }
    out[t] = acc;
  }
}

static __device__ __forceinline__ void relu4(v8f d[4]) {
#pragma unroll
  for (int t = 0; t < 4; ++t)
#pragma unroll
    for (int r = 0; r < 8; ++r) d[t][r] = fmaxf(d[t][r], 0.f);
}

static __device__ __forceinline__ void store_D(float* base, int half,
                                               const v8f d[4]) {
#pragma unroll
  for (int t = 0; t < 4; ++t)
#pragma unroll
    for (int r = 0; r < 8; ++r)
      base[(size_t)(16 * t + 8 * half + r) * NKTOT] = d[t][r];
}

// Park a 4-tile f32 D result in a per-lane LDS buffer (32 contiguous dwords).
static __device__ __forceinline__ void lds_put32(float* p, const v8f d[4]) {
#pragma unroll
  for (int t = 0; t < 4; ++t) {
    float4 a = make_float4(d[t][0], d[t][1], d[t][2], d[t][3]);
    float4 b = make_float4(d[t][4], d[t][5], d[t][6], d[t][7]);
    *(float4*)(p + t * 8)     = a;
    *(float4*)(p + t * 8 + 4) = b;
  }
}

// ---------------------------------------------------------------------------
// Kernel 1: fused key/value/pem/peb/relation/we1/we2 -> weight, value2 to ws
// ---------------------------------------------------------------------------
__global__ void __launch_bounds__(256, 1) k1_fused(
    const float* __restrict__ xyz, const float* __restrict__ feat,
    const float* __restrict__ kW, const float* __restrict__ kb,
    const float* __restrict__ vW, const float* __restrict__ vb,
    const float* __restrict__ pm1W, const float* __restrict__ pm1b,
    const float* __restrict__ pm2W, const float* __restrict__ pm2b,
    const float* __restrict__ pb1W, const float* __restrict__ pb1b,
    const float* __restrict__ pb2W, const float* __restrict__ pb2b,
    const float* __restrict__ we1W, const float* __restrict__ we1b,
    const float* __restrict__ we2W, const float* __restrict__ we2b,
    float* __restrict__ wsW, float* __restrict__ wsV) {
  extern __shared__ __align__(32) char smem[];
  __bf16* ldsW = (__bf16*)smem;                        // 8 convs * 8KB = 64KB
  float*  ldsB = (float*)(smem + 8 * 4096 * 2);        // 8 * 64 f32 = 2KB
  float*  ldsP = (float*)(smem + 8 * 4096 * 2 + 8 * 64 * 4);  // pem: 36KB
  float*  ldsQ = ldsP + 8 * 32 * PSTRIDE;                     // peb: 36KB

  const float* Ws[8]  = {pm1W, pb1W, pm2W, pb2W, kW, vW, we1W, we2W};
  const float* Bsv[8] = {pm1b, pb1b, pm2b, pb2b, kb, vb, we1b, we2b};
  const int    cins[8] = {3, 3, 64, 64, 64, 64, 64, 64};
  stage_weights(ldsW, ldsB, Ws, Bsv, cins, 8);
  __syncthreads();

  enum { PM1 = 0, PB1 = 1, PM2 = 2, PB2 = 3, KC = 4, VC = 5, WE1 = 6, WE2 = 7 };

  const int wid  = (blockIdx.x * blockDim.x + threadIdx.x) >> 5;
  const int wv   = threadIdx.x >> 5;   // wave within block
  const int lane = threadIdx.x & 31;
  const int col  = lane & 15;          // position within tile == k index
  const int half = lane >> 4;

  float* pbase = ldsP + (wv * 32 + lane) * PSTRIDE;
  float* qbase = ldsQ + (wv * 32 + lane) * PSTRIDE;

  for (int it = 0; it < TILES_PER_WAVE; ++it) {
    const int tile = wid + it * WAVES_TOTAL;
    const int p0   = tile * 16;
    const int b    = p0 / NKTOT;
    const int pin  = p0 % NKTOT;

    const float* xyzB  = xyz  + (size_t)b * 3   * NKTOT + pin + col;
    const float* featB = feat + (size_t)b * CHN * NKTOT + pin + col;

    // xyz B-operand: channels 0..2 live in lanes 0-15, rest zero-padded
    v16bf bx;
#pragma unroll
    for (int e = 0; e < 16; ++e) {
      float xv = 0.f;
      if (e < 3 && half == 0) xv = xyzB[(size_t)e * NKTOT];
      bx[e] = (__bf16)xv;
    }

    v8f t4[4], acc4[4];
    v16bf bop[2];

    // pem = pm2 @ relu(pm1 @ xyz)  -> LDS
    conv_apply<1>(ldsW, ldsB, PM1, lane, &bx, t4);
    relu4(t4);
    bop[0] = relayout(t4[0], t4[1], half);
    bop[1] = relayout(t4[2], t4[3], half);
    conv_apply<2>(ldsW, ldsB, PM2, lane, bop, acc4);
    lds_put32(pbase, acc4);

    // peb = pb2 @ relu(pb1 @ xyz)  -> LDS
    conv_apply<1>(ldsW, ldsB, PB1, lane, &bx, t4);
    relu4(t4);
    bop[0] = relayout(t4[0], t4[1], half);
    bop[1] = relayout(t4[2], t4[3], half);
    conv_apply<2>(ldsW, ldsB, PB2, lane, bop, acc4);
    lds_put32(qbase, acc4);

    // grouped_feature B-operand (64 channels x 16 positions)
    v16bf bf[2];
#pragma unroll
    for (int j = 0; j < 2; ++j)
#pragma unroll
      for (int e = 0; e < 16; ++e)
        bf[j][e] = (__bf16)featB[(size_t)(32 * j + 16 * half + e) * NKTOT];

    // value2 = vW@feat + vb + peb  -> ws  (accumulator freed immediately)
    conv_apply<2>(ldsW, ldsB, VC, lane, bf, acc4);
#pragma unroll
    for (int t = 0; t < 4; ++t)
#pragma unroll
      for (int r = 0; r < 8; ++r) acc4[t][r] += qbase[t * 8 + r];
    store_D(wsV + (size_t)b * CHN * NKTOT + pin + col, half, acc4);

    // relation = (kW@feat + kb) * pem + peb   (pem/peb streamed from LDS)
    conv_apply<2>(ldsW, ldsB, KC, lane, bf, t4);
#pragma unroll
    for (int t = 0; t < 4; ++t)
#pragma unroll
      for (int r = 0; r < 8; ++r)
        t4[t][r] = fmaf(t4[t][r], pbase[t * 8 + r], qbase[t * 8 + r]);

    // weight = we2 @ relu(we1 @ relation)  -> ws
    bop[0] = relayout(t4[0], t4[1], half);
    bop[1] = relayout(t4[2], t4[3], half);
    conv_apply<2>(ldsW, ldsB, WE1, lane, bop, acc4);
    relu4(acc4);
    bop[0] = relayout(acc4[0], acc4[1], half);
    bop[1] = relayout(acc4[2], acc4[3], half);
    conv_apply<2>(ldsW, ldsB, WE2, lane, bop, t4);
    store_D(wsW + (size_t)b * CHN * NKTOT + pin + col, half, t4);
  }
}

// ---------------------------------------------------------------------------
// Kernel 2: softmax stats over n (axis 2) per (b, c, k): max, sum(exp)
// One workgroup per (b*C + c) plane of [N][K].
// ---------------------------------------------------------------------------
__global__ void __launch_bounds__(256) k2_softmax(const float* __restrict__ wsW,
                                                  float* __restrict__ wsM,
                                                  float* __restrict__ wsS) {
  const int bc = blockIdx.x;
  const int t  = threadIdx.x;
  const int k  = t & 15;
  const int g  = t >> 4;
  const float* base = wsW + (size_t)bc * NKTOT + k;

  float m = -3.4e38f, s = 0.f;
  for (int n = g; n < NPOINT; n += 16) {
    float x = base[(size_t)n * KNBR];
    if (x > m) { s = s * __expf(m - x) + 1.f; m = x; }
    else       { s += __expf(x - m); }
  }

  __shared__ float sm[256], ss[256];
  sm[t] = m; ss[t] = s;
  __syncthreads();
  for (int off = 128; off >= 16; off >>= 1) {
    if (t < off) {
      float m1 = sm[t], s1 = ss[t], m2 = sm[t + off], s2 = ss[t + off];
      float mm = fmaxf(m1, m2);
      sm[t] = mm;
      ss[t] = s1 * __expf(m1 - mm) + s2 * __expf(m2 - mm);
    }
    __syncthreads();
  }
  if (t < 16) {
    wsM[bc * KNBR + t] = sm[t];
    wsS[bc * KNBR + t] = ss[t];
  }
}

// ---------------------------------------------------------------------------
// Kernel 3: score = softmax(weight); out = resW @ relu(score * value2) + resb
// ---------------------------------------------------------------------------
__global__ void __launch_bounds__(256, 1) k3_out(
    const float* __restrict__ wsW, const float* __restrict__ wsV,
    const float* __restrict__ wsM, const float* __restrict__ wsS,
    const float* __restrict__ resW, const float* __restrict__ resb,
    float* __restrict__ out) {
  extern __shared__ __align__(32) char smem[];
  __bf16* ldsW = (__bf16*)smem;                 // 1 conv * 8KB
  float*  ldsB = (float*)(smem + 4096 * 2);     // 64 f32

  const float* Ws[1]  = {resW};
  const float* Bsv[1] = {resb};
  const int    cins[1] = {64};
  stage_weights(ldsW, ldsB, Ws, Bsv, cins, 1);
  __syncthreads();

  const int wid  = (blockIdx.x * blockDim.x + threadIdx.x) >> 5;
  const int lane = threadIdx.x & 31;
  const int col  = lane & 15;   // == k
  const int half = lane >> 4;

  for (int it = 0; it < TILES_PER_WAVE; ++it) {
    const int tile = wid + it * WAVES_TOTAL;
    const int p0   = tile * 16;
    const int b    = p0 / NKTOT;
    const int pin  = p0 % NKTOT;

    v16bf bop[2];
#pragma unroll
    for (int j = 0; j < 2; ++j)
#pragma unroll
      for (int e = 0; e < 16; ++e) {
        int c = 32 * j + 16 * half + e;
        size_t gi = (size_t)b * CHN * NKTOT + (size_t)c * NKTOT + pin + col;
        int msi = (b * CHN + c) * KNBR + col;
        float sc = __expf(wsW[gi] - wsM[msi]) / wsS[msi];
        float f  = fmaxf(sc * wsV[gi], 0.f);
        bop[j][e] = (__bf16)f;
      }

    v8f o4[4];
    conv_apply<2>(ldsW, ldsB, 0, lane, bop, o4);
    store_D(out + (size_t)b * CHN * NKTOT + pin + col, half, o4);
  }
}

// ---------------------------------------------------------------------------
// Host launch
// ---------------------------------------------------------------------------
extern "C" void kernel_launch(void* const* d_in, const int* in_sizes, int n_in,
                              void* d_out, int out_size, void* d_ws,
                              size_t ws_size, hipStream_t stream) {
  (void)in_sizes; (void)n_in; (void)out_size; (void)ws_size;
  const float* xyz  = (const float*)d_in[0];
  const float* feat = (const float*)d_in[1];
  const float* kW   = (const float*)d_in[2];
  const float* kb   = (const float*)d_in[3];
  const float* vW   = (const float*)d_in[4];
  const float* vb   = (const float*)d_in[5];
  const float* pm1W = (const float*)d_in[6];
  const float* pm1b = (const float*)d_in[7];
  const float* pm2W = (const float*)d_in[8];
  const float* pm2b = (const float*)d_in[9];
  const float* pb1W = (const float*)d_in[10];
  const float* pb1b = (const float*)d_in[11];
  const float* pb2W = (const float*)d_in[12];
  const float* pb2b = (const float*)d_in[13];
  const float* we1W = (const float*)d_in[14];
  const float* we1b = (const float*)d_in[15];
  const float* we2W = (const float*)d_in[16];
  const float* we2b = (const float*)d_in[17];
  const float* resW = (const float*)d_in[18];
  const float* resb = (const float*)d_in[19];

  const size_t plane = (size_t)BATCH * CHN * NKTOT;  // 33.5M f32
  float* wsW = (float*)d_ws;
  float* wsV = wsW + plane;
  float* wsM = wsV + plane;
  float* wsS = wsM + (size_t)BATCH * CHN * KNBR;

  const unsigned smem1 =
      8 * 4096 * 2 + 8 * 64 * 4 + 2 * 8 * 32 * PSTRIDE * 4;  // 141312 B
  const unsigned smem3 = 4096 * 2 + 64 * 4;                  // 8448 B

  k1_fused<<<NBLOCKS, 256, smem1, stream>>>(
      xyz, feat, kW, kb, vW, vb, pm1W, pm1b, pm2W, pm2b, pb1W, pb1b, pb2W,
      pb2b, we1W, we1b, we2W, we2b, wsW, wsV);
  k2_softmax<<<BATCH * CHN, 256, 0, stream>>>(wsW, wsM, wsS);
  k3_out<<<NBLOCKS, 256, smem3, stream>>>(wsW, wsV, wsM, wsS, resW, resb,
                                          (float*)d_out);
}
